// MelDecoder_4303557231105
// MI455X (gfx1250) — matrix operand order repaired and checked by reading, not compiled, not run on hardware
//
#include <hip/hip_runtime.h>

// ---------------------------------------------------------------------------
// Mamba mel-decoder for MI455X (gfx1250), wave32 + WMMA bf16 (split fp32).
//   B=4, L=2048, d_model=512, d_inner=1024, d_state=16, dt_rank=32, conv=4
// GEMMs: fp32 operands split once into hi/lo bf16 at the LDS-staging step,
// stored pre-packed in the WMMA VGPR layout; inner loop is pure ds_load_b32 +
// v_wmma_f32_16x16x32_bf16 (3 products/K-step recover ~16 mantissa bits).
// ---------------------------------------------------------------------------

#define DMODEL  512
#define NLAYER  4
#define NMELS   80
#define DSTATE  16
#define DTRANK  32
#define DINNER  1024
#define SEQ     2048
#define BATCH   4
#define NTOK    (BATCH * SEQ)      // 8192

typedef __attribute__((ext_vector_type(16))) __bf16 v16bf;
typedef __attribute__((ext_vector_type(8)))  float  v8f;

__device__ __forceinline__ float sig_(float x) { return 1.0f / (1.0f + __expf(-x)); }

__device__ __forceinline__ unsigned pack_bf2(__bf16 a, __bf16 b) {
    union { __bf16 h[2]; unsigned u; } p;
    p.h[0] = a; p.h[1] = b;
    return p.u;
}

// ---------------- RMSNorm: one row per wave, wave32 shuffle reduction -------
__global__ __launch_bounds__(256)
void rmsnorm_kernel(const float* __restrict__ x, const float* __restrict__ w,
                    float* __restrict__ out, int nrows)
{
    int wave = blockIdx.x * (blockDim.x >> 5) + (threadIdx.x >> 5);
    if (wave >= nrows) return;
    int lane = threadIdx.x & 31;
    const float* xr = x + (size_t)wave * DMODEL;
    float v[DMODEL / 32];
    float ss = 0.f;
#pragma unroll
    for (int i = 0; i < DMODEL / 32; ++i) {
        float t = xr[lane + i * 32];
        v[i] = t;
        ss += t * t;
    }
#pragma unroll
    for (int off = 16; off >= 1; off >>= 1) ss += __shfl_xor(ss, off, 32);
    float scale = rsqrtf(ss * (1.0f / DMODEL) + 1e-6f);
    float* orow = out + (size_t)wave * DMODEL;
#pragma unroll
    for (int i = 0; i < DMODEL / 32; ++i) {
        int c = lane + i * 32;
        orow[c] = v[i] * scale * w[c];
    }
}

// ---------------- GEMM: C[M,N] (+)= A[M,K] @ B[K,N], split-bf16 WMMA --------
// Block tile 64x64, 8 waves, each wave -> 16x32 (two 16x16 accumulators).
// LDS holds hi/lo bf16 tiles pre-packed as dwords in WMMA fragment order:
//   A: [m][k] row-major, pairs (k,k+1) -> one dword (A-frag VGPR)
//   B: [n][k] TRANSPOSED,  pairs (k,k+1) -> one dword (B-frag VGPR)
#define TM 64
#define TN 64
#define TK 32
#define LSTRIDE 17   // dwords per row (= 34 bf16), odd -> bank-conflict free

__global__ __launch_bounds__(256)
void gemm_wmma(const float* __restrict__ A, int lda,
               const float* __restrict__ B, int ldb,
               float* __restrict__ C, int ldc,
               int M, int N, int K, int accflag)
{
    __shared__ unsigned lsAh[TM * LSTRIDE];   // A hi, 64 rows x 16 dwords
    __shared__ unsigned lsAl[TM * LSTRIDE];   // A lo
    __shared__ unsigned lsBh[TN * LSTRIDE];   // B hi (transposed: n rows, k cols)
    __shared__ unsigned lsBl[TN * LSTRIDE];   // B lo

    const int tid  = threadIdx.x;
    const int m0   = blockIdx.x * TM;
    const int n0   = blockIdx.y * TN;
    const int wv   = tid >> 5;
    const int lane = tid & 31;
    const int mb   = (wv & 3) * 16;     // wave's row tile inside block
    const int nb   = (wv >> 2) * 32;    // wave's col tile inside block
    const int ln   = lane & 15;
    const int hlf  = lane >> 4;

    v8f acc0 = {};
    v8f acc1 = {};

    // cooperative-load mapping
    const int arow = tid >> 2;          // 0..63  (A: one row, 8 k's)
    const int acol = (tid & 3) * 8;     // 0,8,16,24
    const int bn   = tid & 63;          // 0..63  (B: one column n, 8 k's)
    const int bk0  = (tid >> 6) * 8;    // 0,8,16,24
    const int bvalid = (n0 + bn) < N;

    for (int k0 = 0; k0 < K; k0 += TK) {
        __syncthreads();
        // ---- A tile 64x32: float4 loads, split, pack pairs into dwords ----
        const float* ag = A + (size_t)(m0 + arow) * lda + k0 + acol;
        float4 a0 = *(const float4*)(ag);
        float4 a1 = *(const float4*)(ag + 4);
        float fa[8] = { a0.x, a0.y, a0.z, a0.w, a1.x, a1.y, a1.z, a1.w };
        {
            unsigned* dh = &lsAh[arow * LSTRIDE + (acol >> 1)];
            unsigned* dl = &lsAl[arow * LSTRIDE + (acol >> 1)];
#pragma unroll
            for (int j = 0; j < 8; j += 2) {
                __bf16 h0 = (__bf16)fa[j],     h1 = (__bf16)fa[j + 1];
                __bf16 l0 = (__bf16)(fa[j]     - (float)h0);
                __bf16 l1 = (__bf16)(fa[j + 1] - (float)h1);
                dh[j >> 1] = pack_bf2(h0, h1);
                dl[j >> 1] = pack_bf2(l0, l1);
            }
        }
        // ---- B tile 32x64: column-strided loads (coalesced across lanes),
        //      transposed into LDS [n][k] with packed k-pairs ----
        const float* bg = B + (size_t)(k0 + bk0) * ldb + n0 + bn;
        float fb[8];
#pragma unroll
        for (int j = 0; j < 8; ++j)
            fb[j] = bvalid ? bg[(size_t)j * ldb] : 0.0f;
        {
            unsigned* dh = &lsBh[bn * LSTRIDE + (bk0 >> 1)];
            unsigned* dl = &lsBl[bn * LSTRIDE + (bk0 >> 1)];
#pragma unroll
            for (int j = 0; j < 8; j += 2) {
                __bf16 h0 = (__bf16)fb[j],     h1 = (__bf16)fb[j + 1];
                __bf16 l0 = (__bf16)(fb[j]     - (float)h0);
                __bf16 l1 = (__bf16)(fb[j + 1] - (float)h1);
                dh[j >> 1] = pack_bf2(h0, h1);
                dl[j >> 1] = pack_bf2(l0, l1);
            }
        }
        if (k0 + TK < K) {
            __builtin_prefetch(ag + TK, 0, 1);                 // global_prefetch_b8
            __builtin_prefetch(bg + (size_t)TK * ldb, 0, 1);
        }
        __syncthreads();

        // ---- fragments: pure ds_load_b32 gathers, already packed ----
        union Frag { v16bf v; unsigned u[8]; };
        Frag ah, al;
        {
            const int abase = (mb + ln) * LSTRIDE + 8 * hlf;
#pragma unroll
            for (int vv = 0; vv < 8; ++vv) {
                int ui = abase + vv;   // A: k = e + 16*half
                ah.u[vv] = lsAh[ui];
                al.u[vv] = lsAl[ui];
            }
        }
        Frag b0h, b0l, b1h, b1l;
        {
            const int b0base = (nb + ln) * LSTRIDE + 8 * hlf;       // B: k = e + 16*half
            const int b1base = (nb + 16 + ln) * LSTRIDE + 8 * hlf;
#pragma unroll
            for (int vv = 0; vv < 8; ++vv) {
                b0h.u[vv] = lsBh[b0base + vv];
                b0l.u[vv] = lsBl[b0base + vv];
                b1h.u[vv] = lsBh[b1base + vv];
                b1l.u[vv] = lsBl[b1base + vv];
            }
        }
        // D = A*B + C, three products recover ~16 mantissa bits
        acc0 = __builtin_amdgcn_wmma_f32_16x16x32_bf16(false, ah.v, false, b0h.v, (short)0, acc0, false, false);
        acc0 = __builtin_amdgcn_wmma_f32_16x16x32_bf16(false, ah.v, false, b0l.v, (short)0, acc0, false, false);
        acc0 = __builtin_amdgcn_wmma_f32_16x16x32_bf16(false, al.v, false, b0h.v, (short)0, acc0, false, false);
        acc1 = __builtin_amdgcn_wmma_f32_16x16x32_bf16(false, ah.v, false, b1h.v, (short)0, acc1, false, false);
        acc1 = __builtin_amdgcn_wmma_f32_16x16x32_bf16(false, ah.v, false, b1l.v, (short)0, acc1, false, false);
        acc1 = __builtin_amdgcn_wmma_f32_16x16x32_bf16(false, al.v, false, b1h.v, (short)0, acc1, false, false);
    }

    // store: C/D layout — VGPR r holds row m = r + 8*half, col n = lane&15
    const int nA = n0 + nb + ln;
    const int nB = nA + 16;
#pragma unroll
    for (int r = 0; r < 8; ++r) {
        int m = m0 + mb + r + 8 * hlf;
        float* cr = C + (size_t)m * ldc;
        if (nA < N) cr[nA] = accflag ? (cr[nA] + acc0[r]) : acc0[r];
        if (nB < N) cr[nB] = accflag ? (cr[nB] + acc1[r]) : acc1[r];
    }
}

// ---------------- causal depthwise conv (width 4) + SiLU --------------------
__global__ __launch_bounds__(256)
void conv_silu_kernel(const float* __restrict__ xz, const float* __restrict__ cw,
                      const float* __restrict__ cb, float* __restrict__ uc)
{
    int idx = blockIdx.x * blockDim.x + threadIdx.x;
    if (idx >= NTOK * DINNER) return;
    int c  = idx & (DINNER - 1);
    int bt = idx >> 10;
    int t  = bt & (SEQ - 1);
    float w0 = cw[c * 4 + 0], w1 = cw[c * 4 + 1], w2 = cw[c * 4 + 2], w3 = cw[c * 4 + 3];
    const float* ub = xz + (size_t)bt * (2 * DINNER) + c;   // u = xz[:, :DINNER]
    float acc = cb[c] + w3 * ub[0];
    if (t >= 1) acc += w2 * ub[-(ptrdiff_t)(1 * 2 * DINNER)];
    if (t >= 2) acc += w1 * ub[-(ptrdiff_t)(2 * 2 * DINNER)];
    if (t >= 3) acc += w0 * ub[-(ptrdiff_t)(3 * 2 * DINNER)];
    uc[idx] = acc * sig_(acc);
}

// ---------------- dt = softplus(dt + bias) ---------------------------------
__global__ __launch_bounds__(256)
void dt_softplus_kernel(float* __restrict__ dt, const float* __restrict__ dtb)
{
    int idx = blockIdx.x * blockDim.x + threadIdx.x;
    if (idx >= NTOK * DINNER) return;
    int c = idx & (DINNER - 1);
    float v = dt[idx] + dtb[c];
    dt[idx] = (v > 20.0f) ? v : log1pf(__expf(v));
}

// ---------------- selective scan, fused SiLU(z) gate + D skip ---------------
// grid (BATCH, DINNER/256), block 256: one thread per channel, serial over t.
__global__ __launch_bounds__(256)
void scan_kernel(const float* __restrict__ dt, const float* __restrict__ uc,
                 const float* __restrict__ dbl, const float* __restrict__ xz,
                 const float* __restrict__ Alog, const float* __restrict__ Dsk,
                 float* __restrict__ y)
{
    __shared__ float sBC[2 * DSTATE];
    const int b = blockIdx.x;
    const int c = blockIdx.y * 256 + threadIdx.x;
    float Ar[DSTATE], h[DSTATE];
#pragma unroll
    for (int n = 0; n < DSTATE; ++n) {
        Ar[n] = -__expf(Alog[c * DSTATE + n]);
        h[n]  = 0.0f;
    }
    const float Dv = Dsk[c];
    for (int t = 0; t < SEQ; ++t) {
        size_t row = (size_t)b * SEQ + t;
        __syncthreads();
        if (threadIdx.x < 2 * DSTATE)
            sBC[threadIdx.x] = dbl[row * 64 + DTRANK + threadIdx.x];  // B then C
        __syncthreads();
        float dtv = dt[row * DINNER + c];
        float uv  = uc[row * DINNER + c];
        float zv  = xz[row * (2 * DINNER) + DINNER + c];
        float du  = dtv * uv;
        float yv  = 0.0f;
#pragma unroll
        for (int n = 0; n < DSTATE; ++n) {
            float dA = __expf(dtv * Ar[n]);
            h[n] = dA * h[n] + du * sBC[n];
            yv  += h[n] * sBC[DSTATE + n];
        }
        yv += uv * Dv;
        yv *= zv * sig_(zv);
        y[row * DINNER + c] = yv;
    }
}

// ---------------------------------------------------------------------------
extern "C" void kernel_launch(void* const* d_in, const int* in_sizes, int n_in,
                              void* d_out, int out_size, void* d_ws, size_t ws_size,
                              hipStream_t stream)
{
    const float* x          = (const float*)d_in[0];
    const float* norm_w     = (const float*)d_in[1];
    const float* in_proj_w  = (const float*)d_in[2];
    const float* conv_w     = (const float*)d_in[3];
    const float* conv_b     = (const float*)d_in[4];
    const float* x_proj_w   = (const float*)d_in[5];
    const float* dt_proj_w  = (const float*)d_in[6];
    const float* dt_proj_b  = (const float*)d_in[7];
    const float* A_log      = (const float*)d_in[8];
    const float* D_skip     = (const float*)d_in[9];
    const float* out_proj_w = (const float*)d_in[10];
    const float* normf_w    = (const float*)d_in[11];
    const float* head_w     = (const float*)d_in[12];

    // workspace layout (floats); total ~203 MB
    float* ws   = (float*)d_ws;
    float* xbuf = ws;                          // 8192*512
    float* hbuf = ws +  4194304;               // 8192*512
    float* xz   = ws +  8388608;               // 8192*2048
    float* uc   = ws + 25165824;               // 8192*1024
    float* dbl  = ws + 33554432;               // 8192*64
    float* dtb  = ws + 34078720;               // 8192*1024
    float* ybuf = ws + 42467328;               // 8192*1024

    // residual stream starts as the input (never mutate d_in)
    hipMemcpyAsync(xbuf, x, (size_t)NTOK * DMODEL * sizeof(float),
                   hipMemcpyDeviceToDevice, stream);

    const dim3 blk(256);
    for (int l = 0; l < NLAYER; ++l) {
        const float* ipw = in_proj_w  + (size_t)l * DMODEL * 2 * DINNER;
        const float* cw  = conv_w     + (size_t)l * DINNER * 4;
        const float* cb  = conv_b     + (size_t)l * DINNER;
        const float* xpw = x_proj_w   + (size_t)l * DINNER * 64;
        const float* dtw = dt_proj_w  + (size_t)l * DTRANK * DINNER;
        const float* dtbias = dt_proj_b + (size_t)l * DINNER;
        const float* Al  = A_log      + (size_t)l * DINNER * DSTATE;
        const float* Dl  = D_skip     + (size_t)l * DINNER;
        const float* opw = out_proj_w + (size_t)l * DINNER * DMODEL;

        rmsnorm_kernel<<<NTOK / 8, blk, 0, stream>>>(xbuf, norm_w + l * DMODEL, hbuf, NTOK);

        // xz = h @ in_proj_w   (8192 x 2048, K=512)
        gemm_wmma<<<dim3(NTOK / TM, (2 * DINNER) / TN), blk, 0, stream>>>(
            hbuf, DMODEL, ipw, 2 * DINNER, xz, 2 * DINNER, NTOK, 2 * DINNER, DMODEL, 0);

        conv_silu_kernel<<<(NTOK * DINNER) / 256, blk, 0, stream>>>(xz, cw, cb, uc);

        // dbl = uc @ x_proj_w  (8192 x 64, K=1024)
        gemm_wmma<<<dim3(NTOK / TM, 1), blk, 0, stream>>>(
            uc, DINNER, xpw, 64, dbl, 64, NTOK, 64, DINNER, 0);

        // dt = dbl[:, :32] @ dt_proj_w  (8192 x 1024, K=32)
        gemm_wmma<<<dim3(NTOK / TM, DINNER / TN), blk, 0, stream>>>(
            dbl, 64, dtw, DINNER, dtb, DINNER, NTOK, DINNER, DTRANK, 0);

        dt_softplus_kernel<<<(NTOK * DINNER) / 256, blk, 0, stream>>>(dtb, dtbias);

        scan_kernel<<<dim3(BATCH, DINNER / 256), blk, 0, stream>>>(
            dtb, uc, dbl, xz, Al, Dl, ybuf);

        // xbuf += y @ out_proj_w  (8192 x 512, K=1024), residual fused via acc
        gemm_wmma<<<dim3(NTOK / TM, DMODEL / TN), blk, 0, stream>>>(
            ybuf, DINNER, opw, DMODEL, xbuf, DMODEL, NTOK, DMODEL, DINNER, 1);
    }

    // final norm + head: out = rmsnorm(x) @ head_w   (8192 x 80, K=512)
    rmsnorm_kernel<<<NTOK / 8, blk, 0, stream>>>(xbuf, normf_w, hbuf, NTOK);
    gemm_wmma<<<dim3(NTOK / TM, (NMELS + TN - 1) / TN), blk, 0, stream>>>(
        hbuf, DMODEL, head_w, NMELS, (float*)d_out, NMELS, NTOK, NMELS, DMODEL, 0);
}
